// DecoderLayer_4063039062687
// MI455X (gfx1250) — matrix operand order repaired and checked by reading, not compiled, not run on hardware
//
#include <hip/hip_runtime.h>
#include <math.h>

typedef __attribute__((ext_vector_type(16))) __bf16 v16bf;
typedef __attribute__((ext_vector_type(8)))  float  v8f;

struct Pair16 { uint4 lo, hi; };

__device__ __forceinline__ v16bf make_frag(uint4 lo, uint4 hi) {
    Pair16 p{lo, hi};
    return __builtin_bit_cast(v16bf, p);
}

__device__ __forceinline__ unsigned short f2bf(float f) {
    unsigned int u = __float_as_uint(f);
    unsigned int r = u + 0x7FFFu + ((u >> 16) & 1u);
    return (unsigned short)(r >> 16);
}

// ---------------------------------------------------------------------------
// fp32 -> bf16 conversion (no transpose), for activations x
// ---------------------------------------------------------------------------
__global__ __launch_bounds__(256) void cvt_bf16_kernel(const float* __restrict__ in,
                                                       unsigned short* __restrict__ out,
                                                       int n) {
    int i = blockIdx.x * 256 + threadIdx.x;
    if (i < n) out[i] = f2bf(in[i]);
}

// ---------------------------------------------------------------------------
// fp32 [K,N] -> bf16 [N,K] transpose-convert (weights, done once per launch)
// block (32,8), LDS-tiled
// ---------------------------------------------------------------------------
__global__ __launch_bounds__(256) void cvt_tr_kernel(const float* __restrict__ in,
                                                     unsigned short* __restrict__ out,
                                                     int K, int N) {
    __shared__ float tl[32][33];
    const int n0 = blockIdx.x * 32, k0 = blockIdx.y * 32;
    const int tx = threadIdx.x, ty = threadIdx.y;
#pragma unroll
    for (int i = 0; i < 32; i += 8)
        tl[ty + i][tx] = in[(size_t)(k0 + ty + i) * N + n0 + tx];   // coalesced along N
    __syncthreads();
#pragma unroll
    for (int i = 0; i < 32; i += 8)
        out[(size_t)(n0 + ty + i) * K + k0 + tx] = f2bf(tl[tx][ty + i]); // coalesced along K
}

// ---------------------------------------------------------------------------
// V transpose: qkv[b*T+t][1536 + h*64 + hd] (bf16) -> vt[((b*H+h)*64+hd)*T + t]
// grid (T/32, 2, B*H), block (32,8), LDS-tiled
// ---------------------------------------------------------------------------
__global__ __launch_bounds__(256) void vtrans_kernel(const unsigned short* __restrict__ qkv,
                                                     unsigned short* __restrict__ vt,
                                                     int T) {
    __shared__ unsigned short tl[32][33];
    const int bh = blockIdx.z;                // 0..23
    const int b = bh / 12, h = bh % 12;
    const int t0 = blockIdx.x * 32, hd0 = blockIdx.y * 32;
    const int tx = threadIdx.x, ty = threadIdx.y;
    const unsigned short* src = qkv + (size_t)(b * T) * 2304 + 1536 + h * 64;
#pragma unroll
    for (int i = 0; i < 32; i += 8)
        tl[ty + i][tx] = src[(size_t)(t0 + ty + i) * 2304 + hd0 + tx];
    __syncthreads();
    unsigned short* dst = vt + ((size_t)bh * 64 + hd0) * T;
#pragma unroll
    for (int i = 0; i < 32; i += 8)
        dst[(size_t)(ty + i) * T + t0 + tx] = tl[tx][ty + i];
}

// ---------------------------------------------------------------------------
// GEMM: C[M,N] = A[M,K](bf16) * Bt[N,K](bf16, pre-transposed) + bias,
// optional exact GELU. Block 256 (8 waves), tile 64(M) x 128(N), K-chunk 32.
// Global loads for chunk k+1 are software-pipelined across chunk k's WMMAs.
// ---------------------------------------------------------------------------
#define GBM 64
#define GBN 128
#define GBK 32
#define APITCH 40
#define BPITCH 40

__global__ __launch_bounds__(256) void gemm_bias_kernel(
    const unsigned short* __restrict__ A,    // [M,K] bf16 row-major
    const unsigned short* __restrict__ Bt,   // [N,K] bf16 row-major (B transposed)
    const float* __restrict__ bias,          // [N]
    float* __restrict__ outF,                // [M,N] fp32 or nullptr
    unsigned short* __restrict__ outB,       // [M,N] bf16 or nullptr
    int M, int N, int K, int gelu)
{
    __shared__ alignas(16) unsigned short As[GBM * APITCH];   // [m][k]
    __shared__ alignas(16) unsigned short Bs[GBN * BPITCH];   // [n][k]

    const int tid  = threadIdx.x;
    const int wid  = tid >> 5;
    const int lane = tid & 31;
    const int half = lane >> 4;
    const int lrow = lane & 15;
    const int wm   = (wid >> 2) * 32;   // 0,32
    const int wn   = (wid & 3) * 32;    // 0..96
    const int bm0  = blockIdx.y * GBM;
    const int bn0  = blockIdx.x * GBN;

    v8f acc[2][2];
#pragma unroll
    for (int i = 0; i < 2; i++)
#pragma unroll
        for (int j = 0; j < 2; j++)
#pragma unroll
            for (int r = 0; r < 8; r++) acc[i][j][r] = 0.0f;

    // A tile: 64x32 ushort, one uint4 per thread
    const int ar = tid >> 2;          // 0..63
    const int ac = (tid & 3) * 8;     // 0,8,16,24
    // Bt tile: 128x32 ushort, two uint4 per thread along K
    const int br = tid >> 1;          // 0..127
    const int bc = (tid & 1) * 16;    // 0,16

    const unsigned short* Ap  = A  + (size_t)(bm0 + ar) * K + ac;
    const unsigned short* Btp = Bt + (size_t)(bn0 + br) * K + bc;

    uint4 areg  = *(const uint4*)(Ap);
    uint4 breg0 = *(const uint4*)(Btp);
    uint4 breg1 = *(const uint4*)(Btp + 8);

    for (int k0 = 0; k0 < K; k0 += GBK) {
        // ---- stage current chunk from regs ----
        *(uint4*)(&As[ar * APITCH + ac]) = areg;
        *(uint4*)(&Bs[br * BPITCH + bc]) = breg0;
        *(uint4*)(&Bs[br * BPITCH + bc + 8]) = breg1;
        __syncthreads();

        // ---- issue next chunk's global loads (overlap with WMMAs) ----
        if (k0 + GBK < K) {
            areg  = *(const uint4*)(Ap + k0 + GBK);
            breg0 = *(const uint4*)(Btp + k0 + GBK);
            breg1 = *(const uint4*)(Btp + k0 + GBK + 8);
        }

        // ---- fragments + WMMA ----
        v16bf af[2], bfm[2];
#pragma unroll
        for (int im = 0; im < 2; im++) {
            int row = wm + im * 16 + lrow;
            af[im] = make_frag(*(const uint4*)(&As[row * APITCH + half * 8]),
                               *(const uint4*)(&As[row * APITCH + half * 8 + 16]));
        }
#pragma unroll
        for (int in = 0; in < 2; in++) {
            int col = wn + in * 16 + lrow;
            bfm[in] = make_frag(*(const uint4*)(&Bs[col * BPITCH + half * 8]),
                                *(const uint4*)(&Bs[col * BPITCH + half * 8 + 16]));
        }
#pragma unroll
        for (int im = 0; im < 2; im++)
#pragma unroll
            for (int in = 0; in < 2; in++)
                acc[im][in] = __builtin_amdgcn_wmma_f32_16x16x32_bf16(
                    false, af[im], false, bfm[in], (short)0, acc[im][in], false, false);
        __syncthreads();
    }

    // ---- epilogue ----
#pragma unroll
    for (int im = 0; im < 2; im++)
#pragma unroll
        for (int in = 0; in < 2; in++) {
            int gn = bn0 + wn + in * 16 + lrow;
            float bv = bias[gn];
#pragma unroll
            for (int r = 0; r < 8; r++) {
                int gm = bm0 + wm + im * 16 + r + 8 * half;
                float v = acc[im][in][r] + bv;
                if (gelu) v = 0.5f * v * (1.0f + erff(v * 0.70710678118654752f));
                if (outF) outF[(size_t)gm * N + gn] = v;
                if (outB) outB[(size_t)gm * N + gn] = f2bf(v);
            }
        }
}

// ---------------------------------------------------------------------------
// Flash attention: per (b,h), online softmax over T keys. Block = 128 (4
// waves); wave w owns a 16-row Q tile. Q/K fragments and V^T fragments are
// direct contiguous global loads; only the P transpose goes through per-wave
// LDS (wave-local s_wait_dscnt, no block barriers at all).
// ---------------------------------------------------------------------------
__global__ __launch_bounds__(128) void flash_attn_kernel(
    const unsigned short* __restrict__ qkv,   // [B*T, 3*D] bf16
    const unsigned short* __restrict__ vt,    // [B*H*64, T] bf16 (V transposed)
    unsigned short* __restrict__ out,         // [B*T, D]   bf16
    int T)
{
    constexpr int D3 = 2304, Dd = 768, HD = 64, Hh = 12;
    const int b  = blockIdx.z;
    const int h  = blockIdx.y;
    const int qt = blockIdx.x * 64;
    const int tid = threadIdx.x, wid = tid >> 5, lane = tid & 31;
    const int half = lane >> 4, lrow = lane & 15;

    __shared__ alignas(16) unsigned short Ps[4][16 * 40];   // per-wave [m][key]

    const unsigned short* qp  = qkv + (size_t)(b * T) * D3 + h * HD;
    const unsigned short* kp  = qp + Dd;
    const unsigned short* vtb = vt + (size_t)(b * Hh + h) * HD * T;

    // Q fragments (16 rows x hd=64 -> two K=32 A-frags), loaded once
    v16bf qa[2];
    const int qrow = qt + wid * 16 + lrow;
#pragma unroll
    for (int c = 0; c < 2; c++) {
        const unsigned short* qr = qp + (size_t)qrow * D3 + c * 32 + half * 8;
        qa[c] = make_frag(*(const uint4*)(qr), *(const uint4*)(qr + 16));
    }

    v8f oacc[4];
#pragma unroll
    for (int c = 0; c < 4; c++)
#pragma unroll
        for (int r = 0; r < 8; r++) oacc[c][r] = 0.0f;
    float m_[8], l_[8];
#pragma unroll
    for (int r = 0; r < 8; r++) { m_[r] = -INFINITY; l_[r] = 0.0f; }

    unsigned short* Pb = Ps[wid];

    for (int j = 0; j < T; j += 32) {
        // ---- scores S = Q K^T for two 16-key tiles ----
        v8f s[2];
#pragma unroll
        for (int jt = 0; jt < 2; jt++) {
            int key = j + jt * 16 + lrow;
            v8f sc;
#pragma unroll
            for (int r = 0; r < 8; r++) sc[r] = 0.0f;
#pragma unroll
            for (int c = 0; c < 2; c++) {
                const unsigned short* kr = kp + (size_t)key * D3 + c * 32 + half * 8;
                v16bf kb = make_frag(*(const uint4*)(kr), *(const uint4*)(kr + 16));
                sc = __builtin_amdgcn_wmma_f32_16x16x32_bf16(
                    false, qa[c], false, kb, (short)0, sc, false, false);
            }
            s[jt] = sc;
        }

        // ---- online softmax (scale 1/sqrt(64) = 0.125) ----
        float newm[8], rs[8];
#pragma unroll
        for (int r = 0; r < 8; r++) {
            float a = s[0][r] * 0.125f, bb = s[1][r] * 0.125f;
            s[0][r] = a; s[1][r] = bb;
            float mx = fmaxf(a, bb);
            mx = fmaxf(mx, __shfl_xor(mx, 1, 32));
            mx = fmaxf(mx, __shfl_xor(mx, 2, 32));
            mx = fmaxf(mx, __shfl_xor(mx, 4, 32));
            mx = fmaxf(mx, __shfl_xor(mx, 8, 32));
            newm[r] = fmaxf(m_[r], mx);
        }
#pragma unroll
        for (int r = 0; r < 8; r++) {
            float p0 = __expf(s[0][r] - newm[r]);
            float p1 = __expf(s[1][r] - newm[r]);
            s[0][r] = p0; s[1][r] = p1;
            float t = p0 + p1;
            t += __shfl_xor(t, 1, 32);
            t += __shfl_xor(t, 2, 32);
            t += __shfl_xor(t, 4, 32);
            t += __shfl_xor(t, 8, 32);
            rs[r] = t;
        }
#pragma unroll
        for (int r = 0; r < 8; r++) {
            float alpha = __expf(m_[r] - newm[r]);
            l_[r] = l_[r] * alpha + rs[r];
            m_[r] = newm[r];
#pragma unroll
            for (int c = 0; c < 4; c++) oacc[c][r] *= alpha;
        }

        // ---- P (C-layout) -> per-wave LDS -> A-layout fragment ----
#pragma unroll
        for (int r = 0; r < 8; r++) {
            Pb[(r + 8 * half) * 40 + lrow]      = f2bf(s[0][r]);
            Pb[(r + 8 * half) * 40 + 16 + lrow] = f2bf(s[1][r]);
        }
        asm volatile("s_wait_dscnt 0" ::: "memory");   // wave-local transpose fence
        v16bf pa = make_frag(*(const uint4*)(&Pb[lrow * 40 + half * 8]),
                             *(const uint4*)(&Pb[lrow * 40 + half * 8 + 16]));

        // ---- O += P @ V (V^T fragments direct from global) ----
#pragma unroll
        for (int c = 0; c < 4; c++) {
            const unsigned short* vr = vtb + (size_t)(c * 16 + lrow) * T + j + half * 8;
            v16bf vf = make_frag(*(const uint4*)(vr), *(const uint4*)(vr + 16));
            oacc[c] = __builtin_amdgcn_wmma_f32_16x16x32_bf16(
                false, pa, false, vf, (short)0, oacc[c], false, false);
        }
    }

    // ---- normalize and store o (bf16) ----
#pragma unroll
    for (int c = 0; c < 4; c++) {
        int gn = h * HD + c * 16 + lrow;
#pragma unroll
        for (int r = 0; r < 8; r++) {
            int gm = b * T + qt + wid * 16 + r + 8 * half;
            out[(size_t)gm * Dd + gn] = f2bf(oacc[c][r] / l_[r]);
        }
    }
}

// ---------------------------------------------------------------------------
// LayerNorm over last dim (768): one block per row.
// ---------------------------------------------------------------------------
__global__ __launch_bounds__(256) void layernorm_kernel(
    const float* __restrict__ in, const float* __restrict__ g,
    const float* __restrict__ bt, float* __restrict__ outF,
    unsigned short* __restrict__ outB, int Dd)
{
    const int row = blockIdx.x;
    const float* x = in + (size_t)row * Dd;
    __shared__ float red[256];

    float s = 0.f, s2 = 0.f;
    for (int c = threadIdx.x; c < Dd; c += 256) { float v = x[c]; s += v; s2 += v * v; }

    red[threadIdx.x] = s; __syncthreads();
    for (int o = 128; o > 0; o >>= 1) {
        if (threadIdx.x < o) red[threadIdx.x] += red[threadIdx.x + o];
        __syncthreads();
    }
    float mu = red[0] / (float)Dd;
    __syncthreads();
    red[threadIdx.x] = s2; __syncthreads();
    for (int o = 128; o > 0; o >>= 1) {
        if (threadIdx.x < o) red[threadIdx.x] += red[threadIdx.x + o];
        __syncthreads();
    }
    float var = red[0] / (float)Dd - mu * mu;
    float rstd = rsqrtf(var + 1e-5f);

    for (int c = threadIdx.x; c < Dd; c += 256) {
        float v = (x[c] - mu) * rstd * g[c] + bt[c];
        if (outF) outF[(size_t)row * Dd + c] = v;
        if (outB) outB[(size_t)row * Dd + c] = f2bf(v);
    }
}

// ---------------------------------------------------------------------------
// Host launch
// ---------------------------------------------------------------------------
extern "C" void kernel_launch(void* const* d_in, const int* in_sizes, int n_in,
                              void* d_out, int out_size, void* d_ws, size_t ws_size,
                              hipStream_t stream) {
    (void)in_sizes; (void)n_in; (void)out_size; (void)ws_size;
    constexpr int Bn = 2, T = 2048, Dd = 768, F = 3072, Hh = 12;
    constexpr int M = Bn * T;              // 4096

    const float* x     = (const float*)d_in[0];
    const float* w_qkv = (const float*)d_in[1];
    const float* b_qkv = (const float*)d_in[2];
    const float* w_out = (const float*)d_in[3];
    const float* b_out = (const float*)d_in[4];
    const float* w_ff1 = (const float*)d_in[5];
    const float* b_ff1 = (const float*)d_in[6];
    const float* w_ff2 = (const float*)d_in[7];
    const float* b_ff2 = (const float*)d_in[8];
    const float* g1    = (const float*)d_in[9];
    const float* bt1   = (const float*)d_in[10];
    const float* g2    = (const float*)d_in[11];
    const float* bt2   = (const float*)d_in[12];

    char* ws = (char*)d_ws;
    // workspace layout (bytes), with overlap-reuse of dead regions
    unsigned short* xb    = (unsigned short*)(ws + 0);          //  6,291,456  x bf16 ; later x1 bf16
    unsigned short* wqkvT = (unsigned short*)(ws + 6291456);    //  3,538,944  [2304,768]
    unsigned short* woutT = (unsigned short*)(ws + 9830400);    //  1,179,648  [768,768]
    unsigned short* wff1T = (unsigned short*)(ws + 11010048);   //  4,718,592  [3072,768]
    unsigned short* wff2T = (unsigned short*)(ws + 15728640);   //  4,718,592  [768,3072]
    unsigned short* qkvb  = (unsigned short*)(ws + 20447232);   // 18,874,368 ; later hb spans qkvb+ob
    unsigned short* ob    = (unsigned short*)(ws + 39321600);   //  6,291,456
    float*          attnf = (float*)(ws + 45613056);            // 12,582,912 ; later ffnf
    unsigned short* vt    = (unsigned short*)(ws + 58195968);   //  6,291,456  [B*H*64, T]
    unsigned short* x1b   = xb;
    unsigned short* hb    = qkvb;
    float*          ffnf  = attnf;

    dim3 blk256(256), blk128(128), blkT(32, 8);

    // 1) convert x (plain) and weights (transposed) to bf16
    cvt_bf16_kernel<<<(M * Dd + 255) / 256, blk256, 0, stream>>>(x, xb, M * Dd);
    cvt_tr_kernel<<<dim3(3 * Dd / 32, Dd / 32), blkT, 0, stream>>>(w_qkv, wqkvT, Dd, 3 * Dd);
    cvt_tr_kernel<<<dim3(Dd / 32, Dd / 32), blkT, 0, stream>>>(w_out, woutT, Dd, Dd);
    cvt_tr_kernel<<<dim3(F / 32, Dd / 32), blkT, 0, stream>>>(w_ff1, wff1T, Dd, F);
    cvt_tr_kernel<<<dim3(Dd / 32, F / 32), blkT, 0, stream>>>(w_ff2, wff2T, F, Dd);

    // 2) QKV projection -> qkv bf16 [M, 3D]
    gemm_bias_kernel<<<dim3((3 * Dd) / GBN, M / GBM), blk256, 0, stream>>>(
        xb, wqkvT, b_qkv, nullptr, qkvb, M, 3 * Dd, Dd, 0);

    // 3) transpose V heads -> vt [B*H*64, T]
    vtrans_kernel<<<dim3(T / 32, 2, Bn * Hh), blkT, 0, stream>>>(qkvb, vt, T);

    // 4) flash attention -> o bf16 [M, D]
    flash_attn_kernel<<<dim3(T / 64, Hh, Bn), blk128, 0, stream>>>(qkvb, vt, ob, T);

    // 5) output projection -> attn_out fp32
    gemm_bias_kernel<<<dim3(Dd / GBN, M / GBM), blk256, 0, stream>>>(
        ob, woutT, b_out, attnf, nullptr, M, Dd, Dd, 0);

    // 6) layernorm1 -> x1 bf16
    layernorm_kernel<<<M, blk256, 0, stream>>>(attnf, g1, bt1, nullptr, x1b, Dd);

    // 7) FF1 + GELU -> h bf16 [M, F]
    gemm_bias_kernel<<<dim3(F / GBN, M / GBM), blk256, 0, stream>>>(
        x1b, wff1T, b_ff1, nullptr, hb, M, F, Dd, 1);

    // 8) FF2 -> ffn_out fp32
    gemm_bias_kernel<<<dim3(Dd / GBN, M / GBM), blk256, 0, stream>>>(
        hb, wff2T, b_ff2, ffnf, nullptr, M, Dd, F, 0);

    // 9) layernorm2 -> d_out fp32
    layernorm_kernel<<<M, blk256, 0, stream>>>(ffnf, g2, bt2, (float*)d_out, nullptr, Dd);
}
